// SillyLinear_54657753809084
// MI455X (gfx1250) — matrix operand order repaired and checked by reading, not compiled, not run on hardware
//
#include <hip/hip_runtime.h>

typedef _Float16 h4  __attribute__((ext_vector_type(4)));
typedef _Float16 h8  __attribute__((ext_vector_type(8)));
typedef _Float16 h16 __attribute__((ext_vector_type(16)));
typedef float    f4  __attribute__((ext_vector_type(4)));
typedef float    f8  __attribute__((ext_vector_type(8)));
typedef unsigned int u32x4 __attribute__((ext_vector_type(4)));
typedef int          i32x4 __attribute__((ext_vector_type(4)));
typedef int          i32x8 __attribute__((ext_vector_type(8)));

#define IN_F    1024
#define OUT_F   1024
#define INT_DIM 128
#define M_TOTAL (8 * 2048)   // 16384 rows

// LDS chunk geometry: 64 N-rows x 128 K-halves, padded +8 halves/row
#define CHUNK_K     128                  // halves of K per staged chunk
#define ROW_PITCH   (CHUNK_K + 8)        // 136 halves = 272B -> 4-bank rotate/row
#define BUF_HALVES  (64 * ROW_PITCH)     // 8704 halves = 17408 B per buffer
#define SMEM_BYTES  (2 * BUF_HALVES * sizeof(_Float16))
#define N_CHUNKS    (IN_F / CHUNK_K)     // 8

// ---------------------------------------------------------------------------
// Kernel 1: W[o,i] = sum_k coef[k] * P_w[k, o*IN_F + i], stored as f16.
// Memory-bound (512 MB stream of P_w); coef[k] is wave-uniform -> scalar load.
// ---------------------------------------------------------------------------
__global__ __launch_bounds__(256) void materialize_w_f16(
    const float* __restrict__ coef, const float* __restrict__ Pw,
    _Float16* __restrict__ Wh) {
  const size_t j0 = ((size_t)blockIdx.x * blockDim.x + threadIdx.x) * 4;
  f4 acc = {0.f, 0.f, 0.f, 0.f};
#pragma unroll 8
  for (int k = 0; k < INT_DIM; ++k) {
    const float c = coef[k];
    const f4 p = *(const f4*)(Pw + (size_t)k * (OUT_F * IN_F) + j0);
    acc += c * p;
  }
  h4 o;
  o[0] = (_Float16)acc[0]; o[1] = (_Float16)acc[1];
  o[2] = (_Float16)acc[2]; o[3] = (_Float16)acc[3];
  *(h4*)(Wh + j0) = o;
}

// ---------------------------------------------------------------------------
// Kernel 2: bias[o] = sum_k bcoef[k] * P_b[k, o]  (tiny; kept in f32)
// ---------------------------------------------------------------------------
__global__ __launch_bounds__(256) void materialize_bias(
    const float* __restrict__ bcoef, const float* __restrict__ Pb,
    float* __restrict__ bias) {
  const int j = blockIdx.x * blockDim.x + threadIdx.x;
  if (j < OUT_F) {
    float acc = 0.f;
#pragma unroll 8
    for (int k = 0; k < INT_DIM; ++k) acc += bcoef[k] * Pb[(size_t)k * OUT_F + j];
    bias[j] = acc;
  }
}

// ---------------------------------------------------------------------------
// TDM: DMA a 2D tile (64 rows x CHUNK_K halves) of row-major f16 W into LDS.
// D# packing per CDNA5 ISA §8.3/8.4:
//   group0: count=1 | lds_addr | global_addr[56:0] | type=2
//   group1: data_size=2B, pad_enable, pad_interval=64 DW (one 128-half row),
//           pad_amount=4 DW (8 halves) -> LDS row pitch 136 halves,
//           tensor_dim0=1024, tensor_dim1=64, tile_dim0=128, tile_dim1=64,
//           tensor_dim0_stride=1024
//   groups 2/3: zero (2D tile)
// ---------------------------------------------------------------------------
__device__ __forceinline__ void tdm_load_w_chunk(unsigned lds_off,
                                                 const _Float16* gptr) {
  const unsigned long long ga = (unsigned long long)gptr;
  u32x4 g0;
  g0[0] = 1u;                                   // count=1 (valid user D#)
  g0[1] = lds_off;                              // lds_addr (bytes)
  g0[2] = (unsigned)(ga & 0xffffffffu);         // global_addr[31:0]
  g0[3] = (unsigned)((ga >> 32) & 0x01ffffffu)  // global_addr[56:32]
        | (2u << 30);                           // type=2 ("image")
  i32x8 g1;
  g1[0] = (1 << 16)                             // data_size = 2 bytes
        | (1 << 20)                             // pad_enable
        | (5 << 22)                             // pad_interval: 64 DWORDs
        | (3 << 25);                            // pad_amount: 4 DWORDs
  g1[1] = (int)((unsigned)IN_F << 16);          // tensor_dim0[15:0] @ bits 63:48
  g1[2] = (int)(64u << 16);                     // tensor_dim1[15:0] @ bits 95:80
  g1[3] = (int)((unsigned)CHUNK_K << 16);       // tile_dim0 @ bits 127:112
  g1[4] = 64;                                   // tile_dim1 @ bits 143:128
  g1[5] = IN_F;                                 // tensor_dim0_stride[31:0]
  g1[6] = 0;
  g1[7] = 0;
  i32x4 gz = {0, 0, 0, 0};
#if defined(__clang_major__) && (__clang_major__ >= 23)
  i32x8 gz8 = {0, 0, 0, 0, 0, 0, 0, 0};
  __builtin_amdgcn_tensor_load_to_lds(g0, g1, gz, gz, gz8, 0);
#else
  __builtin_amdgcn_tensor_load_to_lds(g0, g1, gz, gz, 0);
#endif
}

// ---------------------------------------------------------------------------
// Kernel 3: y[m,n] = sum_k x[m,k] * W[n,k] + bias[n]
// Block = 256 threads = 8 waves; block tile 128 M x 64 N.
// Wave w owns M-tile (8*blockIdx.x + w) and all four 16-wide N-tiles.
// W slab staged into LDS by the Tensor Data Mover, double buffered per
// 128-K-half chunk; B fragments come from LDS (ds_load_b128, conflict-free
// via the 136-half row pitch), A fragments stream from global f32 x with
// on-the-fly cvt to f16.
// ---------------------------------------------------------------------------
__global__ __launch_bounds__(256) void wmma_gemm_bias(
    const float* __restrict__ X, const _Float16* __restrict__ Wh,
    const float* __restrict__ bias, float* __restrict__ Y) {
  extern __shared__ _Float16 smem[];            // 2 x BUF_HALVES

  const int lane    = threadIdx.x & 31;
  const int wave    = threadIdx.x >> 5;
  const int lane16  = lane & 15;
  const int halfSel = lane >> 4;                // 0 or 1

  const int mTile = blockIdx.x * 8 + wave;      // 16-row tile index
  const int nBase = blockIdx.y * 64;            // first of 4 N-tiles

  // A source: row m of x, at the lane's K-phase (A frag ISA layout)
  const int m = mTile * 16 + lane16;
  const float* xrow = X + (size_t)m * IN_F + halfSel * 8;

  // B source rows in LDS: row (t*16 + lane16), K-phase halfSel*8
  const int rowOff0 = (lane16 +  0) * ROW_PITCH + halfSel * 8;
  const int rowOff1 = (lane16 + 16) * ROW_PITCH + halfSel * 8;
  const int rowOff2 = (lane16 + 32) * ROW_PITCH + halfSel * 8;
  const int rowOff3 = (lane16 + 48) * ROW_PITCH + halfSel * 8;

  f8 c0 = {}, c1 = {}, c2 = {}, c3 = {};

  // W slab global base for this block's 64 N-rows
  const _Float16* wslab = Wh + (size_t)nBase * IN_F;

  // Prologue: DMA chunk 0 into buffer 0 (wave 0 drives the TDM)
  if (wave == 0) tdm_load_w_chunk(0u, wslab);

#define COMPUTE_CHUNK(cidx)                                                   \
  {                                                                           \
    const _Float16* sb = smem + ((cidx) & 1) * BUF_HALVES;                    \
    _Pragma("unroll")                                                         \
    for (int ks = 0; ks < CHUNK_K / 32; ++ks) {                               \
      const int kg = (cidx) * CHUNK_K + ks * 32;                              \
      const f4 a0 = *(const f4*)(xrow + kg);                                  \
      const f4 a1 = *(const f4*)(xrow + kg + 4);                              \
      const f4 a2 = *(const f4*)(xrow + kg + 16);                             \
      const f4 a3 = *(const f4*)(xrow + kg + 20);                             \
      h16 A;                                                                  \
      A[0]  = (_Float16)a0[0]; A[1]  = (_Float16)a0[1];                       \
      A[2]  = (_Float16)a0[2]; A[3]  = (_Float16)a0[3];                       \
      A[4]  = (_Float16)a1[0]; A[5]  = (_Float16)a1[1];                       \
      A[6]  = (_Float16)a1[2]; A[7]  = (_Float16)a1[3];                       \
      A[8]  = (_Float16)a2[0]; A[9]  = (_Float16)a2[1];                       \
      A[10] = (_Float16)a2[2]; A[11] = (_Float16)a2[3];                       \
      A[12] = (_Float16)a3[0]; A[13] = (_Float16)a3[1];                       \
      A[14] = (_Float16)a3[2]; A[15] = (_Float16)a3[3];                       \
      const int kl = ks * 32;                                                 \
      const h8 b0l = *(const h8*)(sb + rowOff0 + kl);                         \
      const h8 b0h = *(const h8*)(sb + rowOff0 + kl + 16);                    \
      const h8 b1l = *(const h8*)(sb + rowOff1 + kl);                         \
      const h8 b1h = *(const h8*)(sb + rowOff1 + kl + 16);                    \
      const h8 b2l = *(const h8*)(sb + rowOff2 + kl);                         \
      const h8 b2h = *(const h8*)(sb + rowOff2 + kl + 16);                    \
      const h8 b3l = *(const h8*)(sb + rowOff3 + kl);                         \
      const h8 b3h = *(const h8*)(sb + rowOff3 + kl + 16);                    \
      const h16 B0 = __builtin_shufflevector(b0l, b0h, 0,1,2,3,4,5,6,7,8,9,10,11,12,13,14,15); \
      const h16 B1 = __builtin_shufflevector(b1l, b1h, 0,1,2,3,4,5,6,7,8,9,10,11,12,13,14,15); \
      const h16 B2 = __builtin_shufflevector(b2l, b2h, 0,1,2,3,4,5,6,7,8,9,10,11,12,13,14,15); \
      const h16 B3 = __builtin_shufflevector(b3l, b3h, 0,1,2,3,4,5,6,7,8,9,10,11,12,13,14,15); \
      c0 = __builtin_amdgcn_wmma_f32_16x16x32_f16(false, A, false, B0, (short)0, c0, false, false); \
      c1 = __builtin_amdgcn_wmma_f32_16x16x32_f16(false, A, false, B1, (short)0, c1, false, false); \
      c2 = __builtin_amdgcn_wmma_f32_16x16x32_f16(false, A, false, B2, (short)0, c2, false, false); \
      c3 = __builtin_amdgcn_wmma_f32_16x16x32_f16(false, A, false, B3, (short)0, c3, false, false); \
    }                                                                         \
  }

  // Steady state: DMA chunk c+1 into the other buffer, wait for chunk c
  // (TDM ops from one wave complete in order -> tensorcnt<=1 means chunk c
  // has landed), then barrier-release all waves to compute on it.
#pragma unroll 2
  for (int c = 0; c < N_CHUNKS - 1; ++c) {
    if (wave == 0) {
      tdm_load_w_chunk(((c + 1) & 1) * (unsigned)(BUF_HALVES * sizeof(_Float16)),
                       wslab + (c + 1) * CHUNK_K);
      __builtin_amdgcn_s_wait_tensorcnt(1);
    }
    __syncthreads();
    COMPUTE_CHUNK(c);
    __syncthreads();   // all readers done before buffer (c&1) is overwritten
  }
  if (wave == 0) __builtin_amdgcn_s_wait_tensorcnt(0);
  __syncthreads();
  COMPUTE_CHUNK(N_CHUNKS - 1);
#undef COMPUTE_CHUNK

  // Epilogue: C/D layout -> M = r + 8*halfSel, N = lane16 (per N-tile)
  const int rowBase = mTile * 16 + 8 * halfSel;
  const float bv0 = bias[nBase +  0 + lane16];
  const float bv1 = bias[nBase + 16 + lane16];
  const float bv2 = bias[nBase + 32 + lane16];
  const float bv3 = bias[nBase + 48 + lane16];
#pragma unroll
  for (int r = 0; r < 8; ++r) {
    float* yr = Y + (size_t)(rowBase + r) * OUT_F + nBase + lane16;
    yr[0]  = c0[r] + bv0;
    yr[16] = c1[r] + bv1;
    yr[32] = c2[r] + bv2;
    yr[48] = c3[r] + bv3;
  }
}

// ---------------------------------------------------------------------------
// Host-side launcher
// ---------------------------------------------------------------------------
extern "C" void kernel_launch(void* const* d_in, const int* in_sizes, int n_in,
                              void* d_out, int out_size, void* d_ws, size_t ws_size,
                              hipStream_t stream) {
  const float* x     = (const float*)d_in[0];   // [8,2048,1024]
  const float* wcoef = (const float*)d_in[1];   // [128]
  const float* bcoef = (const float*)d_in[2];   // [128]
  const float* Pw    = (const float*)d_in[3];   // [128, 1024*1024]
  const float* Pb    = (const float*)d_in[4];   // [128, 1024]
  float* out = (float*)d_out;                   // [8,2048,1024]

  // Workspace layout: [0, 2MB) Wh (f16 1024x1024), then bias f32[1024]
  _Float16* Wh  = (_Float16*)d_ws;
  float* bias   = (float*)((char*)d_ws + (size_t)OUT_F * IN_F * sizeof(_Float16));

  materialize_w_f16<<<(OUT_F * IN_F) / (256 * 4), 256, 0, stream>>>(wcoef, Pw, Wh);
  materialize_bias<<<(OUT_F + 255) / 256, 256, 0, stream>>>(bcoef, Pb, bias);

  dim3 grid(M_TOTAL / 128, OUT_F / 64);
  wmma_gemm_bias<<<grid, 256, SMEM_BYTES, stream>>>(x, Wh, bias, out);
}
